// GraphAttention_2147483648071
// MI455X (gfx1250) — matrix operand order repaired and checked by reading, not compiled, not run on hardware
//
#include <hip/hip_runtime.h>
#include <hip/hip_bf16.h>

// ---------------------------------------------------------------------------
// GAT forward for MI455X (gfx1250, wave32, WMMA f32_16x16x32_f16)
// N=4096 nodes, F_IN=512, HEADS=8, UNITS=64
// ---------------------------------------------------------------------------

typedef _Float16 half_t;
typedef __attribute__((ext_vector_type(16))) _Float16 v16h;
typedef __attribute__((ext_vector_type(8)))  _Float16 v8h;
typedef __attribute__((ext_vector_type(8)))  float    v8f;

#define GAT_N     4096
#define GAT_F     512
#define GAT_U     64
#define GAT_H     8

// ---------------------------------------------------------------------------
// Kernel 0: W [H,F,U] f32  ->  Wt [H,U,F] f16   (B-fragment friendly layout)
// ---------------------------------------------------------------------------
__global__ void gat_wt(const float* __restrict__ W, half_t* __restrict__ Wt) {
    int t = blockIdx.x * 256 + threadIdx.x;          // t < H*F*U
    int h = t / (GAT_F * GAT_U);
    int rem = t - h * (GAT_F * GAT_U);
    int f = rem / GAT_U;
    int u = rem - f * GAT_U;
    Wt[((size_t)(h * GAT_U + u)) * GAT_F + f] = (half_t)W[t];
}

// ---------------------------------------------------------------------------
// Kernel 1: feats = X @ W per head, stored transposed as f16:
//   featsT[H, U, N]  (so GEMM2 B-fragments are contiguous per lane)
// one wave per 16x16 output tile; K loop over 512 in steps of 32
// ---------------------------------------------------------------------------
__global__ void gat_feats(const float* __restrict__ X,
                          const half_t* __restrict__ Wt,
                          half_t* __restrict__ featsT) {
    int l = threadIdx.x;          // 0..31 (one wave per block)
    int r = l & 15;               // lane % 16
    int hf = l >> 4;              // lane half (0/1)
    int m0 = blockIdx.x * 16;     // node rows
    int u0 = blockIdx.y * 16;     // unit cols
    int h  = blockIdx.z;          // head

    v8f acc = {};
    const float*  xrow = X + (size_t)(m0 + r) * GAT_F;
    const half_t* wrow = Wt + ((size_t)(h * GAT_U + u0 + r)) * GAT_F + 8 * hf;

    for (int k0 = 0; k0 < GAT_F; k0 += 32) {
        // ---- A fragment (16x32 f16): lane r holds row m0+r,
        //      K = k0 + 8*hf + [0..7]  and  k0 + 16 + 8*hf + [0..7]
        const float4* p1 = (const float4*)(xrow + k0 + 8 * hf);
        const float4* p2 = (const float4*)(xrow + k0 + 16 + 8 * hf);
        float4 x0 = p1[0], x1 = p1[1];
        float4 x2 = p2[0], x3 = p2[1];
        v16h a;
        a[0]  = (half_t)x0.x; a[1]  = (half_t)x0.y; a[2]  = (half_t)x0.z; a[3]  = (half_t)x0.w;
        a[4]  = (half_t)x1.x; a[5]  = (half_t)x1.y; a[6]  = (half_t)x1.z; a[7]  = (half_t)x1.w;
        a[8]  = (half_t)x2.x; a[9]  = (half_t)x2.y; a[10] = (half_t)x2.z; a[11] = (half_t)x2.w;
        a[12] = (half_t)x3.x; a[13] = (half_t)x3.y; a[14] = (half_t)x3.z; a[15] = (half_t)x3.w;

        // ---- B fragment (32x16 f16): lane r holds col u0+r,
        //      K = k0 + 8*hf + [0..7]  and  k0 + 16 + 8*hf + [0..7]
        const v8h* bp = (const v8h*)(wrow + k0);
        v8h b0 = bp[0], b1 = bp[2];
        v16h b;
#pragma unroll
        for (int e = 0; e < 8; ++e) { b[e] = b0[e]; b[8 + e] = b1[e]; }

        acc = __builtin_amdgcn_wmma_f32_16x16x32_f16(
            false, a, false, b, (short)0, acc, false, false);
    }

    // C/D layout: lane l, VGPR v -> (M = v + 8*hf, Ncol = r); store transposed
#pragma unroll
    for (int v = 0; v < 8; ++v) {
        int n = m0 + v + 8 * hf;
        int u = u0 + r;
        featsT[((size_t)(h * GAT_U + u)) * GAT_N + n] = (half_t)acc[v];
    }
}

// ---------------------------------------------------------------------------
// Kernel 2: e_self[h,n] = feats[h,n,:].a_self[h], e_neigh likewise (f32)
// ---------------------------------------------------------------------------
__global__ void gat_escore(const half_t* __restrict__ featsT,
                           const float* __restrict__ a_self,
                           const float* __restrict__ a_neigh,
                           float* __restrict__ e_self,
                           float* __restrict__ e_neigh) {
    int t = blockIdx.x * 256 + threadIdx.x;   // t < H*N
    int h = t >> 12;
    int n = t & (GAT_N - 1);
    float es = 0.f, en = 0.f;
#pragma unroll 8
    for (int u = 0; u < GAT_U; ++u) {
        float v = (float)featsT[((size_t)(h * GAT_U + u)) * GAT_N + n];
        es += v * a_self[h * GAT_U + u];
        en += v * a_neigh[h * GAT_U + u];
    }
    e_self[t] = es;
    e_neigh[t] = en;
}

// ---------------------------------------------------------------------------
// Kernel 2b: per-head max of e_neigh (for a safe softmax upper bound)
// ---------------------------------------------------------------------------
__global__ void gat_emax(const float* __restrict__ e_neigh,
                         float* __restrict__ mhead) {
    __shared__ float red[256];
    int h = blockIdx.x;
    float mx = -1e30f;
    for (int n = threadIdx.x; n < GAT_N; n += 256)
        mx = fmaxf(mx, e_neigh[h * GAT_N + n]);
    red[threadIdx.x] = mx;
    __syncthreads();
    for (int s = 128; s > 0; s >>= 1) {
        if (threadIdx.x < s) red[threadIdx.x] = fmaxf(red[threadIdx.x], red[threadIdx.x + s]);
        __syncthreads();
    }
    if (threadIdx.x == 0) mhead[h] = red[0];
}

// ---------------------------------------------------------------------------
// Kernel 3: fused masked-softmax attention + aggregation.
// grid = (N/16 row-blocks, HEADS); block = 128 (4 waves).
// Wave w handles j-chunks {w*32, w*32+128, ...}; each chunk:
//   build P fragment (16 rows x 32 cols of exp(score - m_i), f16),
//   4x WMMA against featsT tiles (U=64), accumulate denominator.
// Cross-wave LDS reduction, then relu(num/den + bias) -> out[N, H*U].
// ---------------------------------------------------------------------------
__device__ __forceinline__ float gat_prob(float av, float ev, float es,
                                          float mi, float& den) {
    float s = es + ev;
    s = (s > 0.f) ? s : 0.2f * s;          // leaky_relu(0.2)
    float p = (av != 0.f) ? __expf(s - mi) : 0.f;  // mask -> exact 0
    den += p;
    return p;
}

__global__ void gat_attn(const float* __restrict__ A,
                         const half_t* __restrict__ featsT,
                         const float* __restrict__ e_self,
                         const float* __restrict__ e_neigh,
                         const float* __restrict__ mhead,
                         const float* __restrict__ bias,
                         float* __restrict__ out) {
    __shared__ float lnum[4][16][64];
    __shared__ float lden[4][2][16];

    int l  = threadIdx.x & 31;
    int w  = threadIdx.x >> 5;
    int r  = l & 15;
    int hf = l >> 4;
    int i0 = blockIdx.x * 16;
    int h  = blockIdx.y;
    int row = i0 + r;

    float es = e_self[h * GAT_N + row];
    float sm = es + mhead[h];
    float mi = (sm > 0.f) ? sm : 0.2f * sm;   // row-wise upper bound on scores

    v8f acc0 = {}, acc1 = {}, acc2 = {}, acc3 = {};
    float den = 0.f;

    const float*  arow = A + (size_t)row * GAT_N;
    const float*  enh  = e_neigh + h * GAT_N;
    const half_t* fcol = featsT + ((size_t)(h * GAT_U + r)) * GAT_N + 8 * hf;

    for (int jb = w * 32; jb < GAT_N; jb += 128) {
        int c1 = jb + 8 * hf;
        int c2 = c1 + 16;
        float4 a0 = *(const float4*)(arow + c1);
        float4 a1 = *(const float4*)(arow + c1 + 4);
        float4 a2 = *(const float4*)(arow + c2);
        float4 a3 = *(const float4*)(arow + c2 + 4);
        float4 e0 = *(const float4*)(enh + c1);
        float4 e1 = *(const float4*)(enh + c1 + 4);
        float4 e2 = *(const float4*)(enh + c2);
        float4 e3 = *(const float4*)(enh + c2 + 4);

        v16h P;
        P[0]  = (half_t)gat_prob(a0.x, e0.x, es, mi, den);
        P[1]  = (half_t)gat_prob(a0.y, e0.y, es, mi, den);
        P[2]  = (half_t)gat_prob(a0.z, e0.z, es, mi, den);
        P[3]  = (half_t)gat_prob(a0.w, e0.w, es, mi, den);
        P[4]  = (half_t)gat_prob(a1.x, e1.x, es, mi, den);
        P[5]  = (half_t)gat_prob(a1.y, e1.y, es, mi, den);
        P[6]  = (half_t)gat_prob(a1.z, e1.z, es, mi, den);
        P[7]  = (half_t)gat_prob(a1.w, e1.w, es, mi, den);
        P[8]  = (half_t)gat_prob(a2.x, e2.x, es, mi, den);
        P[9]  = (half_t)gat_prob(a2.y, e2.y, es, mi, den);
        P[10] = (half_t)gat_prob(a2.z, e2.z, es, mi, den);
        P[11] = (half_t)gat_prob(a2.w, e2.w, es, mi, den);
        P[12] = (half_t)gat_prob(a3.x, e3.x, es, mi, den);
        P[13] = (half_t)gat_prob(a3.y, e3.y, es, mi, den);
        P[14] = (half_t)gat_prob(a3.z, e3.z, es, mi, den);
        P[15] = (half_t)gat_prob(a3.w, e3.w, es, mi, den);

        // 4 B-fragments: featsT tile [jb..jb+31] x [t*16..t*16+15]
#pragma unroll
        for (int t = 0; t < 4; ++t) {
            const v8h* bp = (const v8h*)(fcol + (size_t)t * 16 * GAT_N + jb);
            v8h b0 = bp[0], b1 = bp[2];
            v16h b;
#pragma unroll
            for (int e = 0; e < 8; ++e) { b[e] = b0[e]; b[8 + e] = b1[e]; }
            if (t == 0) acc0 = __builtin_amdgcn_wmma_f32_16x16x32_f16(false, P, false, b, (short)0, acc0, false, false);
            if (t == 1) acc1 = __builtin_amdgcn_wmma_f32_16x16x32_f16(false, P, false, b, (short)0, acc1, false, false);
            if (t == 2) acc2 = __builtin_amdgcn_wmma_f32_16x16x32_f16(false, P, false, b, (short)0, acc2, false, false);
            if (t == 3) acc3 = __builtin_amdgcn_wmma_f32_16x16x32_f16(false, P, false, b, (short)0, acc3, false, false);
        }
    }

    // spill per-wave partials to LDS (C layout: M = v + 8*hf, Ncol = r)
#pragma unroll
    for (int v = 0; v < 8; ++v) {
        int m = v + 8 * hf;
        lnum[w][m][r]      = acc0[v];
        lnum[w][m][16 + r] = acc1[v];
        lnum[w][m][32 + r] = acc2[v];
        lnum[w][m][48 + r] = acc3[v];
    }
    lden[w][hf][r] = den;
    __syncthreads();

    // final reduce: 16x64 outputs over 128 threads (8 each)
#pragma unroll
    for (int q = 0; q < 8; ++q) {
        int idx = threadIdx.x + q * 128;   // 0..1023
        int rr = idx >> 6;
        int uu = idx & 63;
        float s = lnum[0][rr][uu] + lnum[1][rr][uu] + lnum[2][rr][uu] + lnum[3][rr][uu];
        float d = lden[0][0][rr] + lden[0][1][rr] + lden[1][0][rr] + lden[1][1][rr]
                + lden[2][0][rr] + lden[2][1][rr] + lden[3][0][rr] + lden[3][1][rr];
        float o = s / d + bias[h * GAT_U + uu];
        out[(size_t)(i0 + rr) * (GAT_H * GAT_U) + h * GAT_U + uu] = (o > 0.f) ? o : 0.f;
    }
}

// ---------------------------------------------------------------------------
// Launcher
// ---------------------------------------------------------------------------
extern "C" void kernel_launch(void* const* d_in, const int* in_sizes, int n_in,
                              void* d_out, int out_size, void* d_ws, size_t ws_size,
                              hipStream_t stream) {
    const float* X       = (const float*)d_in[0];   // [1,N,F]
    const float* A       = (const float*)d_in[1];   // [1,N,N]
    const float* W       = (const float*)d_in[2];   // [H,F,U]
    const float* b       = (const float*)d_in[3];   // [H,U]
    const float* a_self  = (const float*)d_in[4];   // [H,U]
    const float* a_neigh = (const float*)d_in[5];   // [H,U]
    float* out           = (float*)d_out;           // [N, H*U]

    // workspace layout
    char* ws = (char*)d_ws;
    half_t* Wt     = (half_t*)ws;                                   // H*U*F f16  = 512 KB
    half_t* featsT = (half_t*)(ws + (size_t)GAT_H * GAT_U * GAT_F * 2);      // H*U*N f16 = 4 MB
    float*  e_self = (float*)((char*)featsT + (size_t)GAT_H * GAT_U * GAT_N * 2);
    float*  e_neigh = e_self + GAT_H * GAT_N;
    float*  mhead   = e_neigh + GAT_H * GAT_N;

    // 0) transpose + f16-convert W
    gat_wt<<<(GAT_H * GAT_F * GAT_U) / 256, 256, 0, stream>>>(W, Wt);

    // 1) feats GEMM (WMMA), stored transposed f16
    gat_feats<<<dim3(GAT_N / 16, GAT_U / 16, GAT_H), 32, 0, stream>>>(X, Wt, featsT);

    // 2) attention coefficient vectors + per-head max
    gat_escore<<<(GAT_H * GAT_N) / 256, 256, 0, stream>>>(featsT, a_self, a_neigh,
                                                          e_self, e_neigh);
    gat_emax<<<GAT_H, 256, 0, stream>>>(e_neigh, mhead);

    // 3) fused masked softmax + aggregation (WMMA)
    gat_attn<<<dim3(GAT_N / 16, GAT_H), 128, 0, stream>>>(A, featsT, e_self, e_neigh,
                                                          mhead, b, out);
}
